// MultiHeadAttention_111669150206
// MI455X (gfx1250) — matrix operand order repaired and checked by reading, not compiled, not run on hardware
//
#include <hip/hip_runtime.h>
#include <hip/hip_bf16.h>

// ---------------------------------------------------------------------------
// Types for CDNA5 WMMA (wave32): v_wmma_f32_16x16x32_bf16
// ---------------------------------------------------------------------------
typedef __attribute__((ext_vector_type(16))) __bf16 v16bf;
typedef __attribute__((ext_vector_type(8)))  __bf16 v8bf;
typedef __attribute__((ext_vector_type(8)))  float  v8f;

#define D_MODEL 1024
#define NUM_HEADS 16
#define D_K 64
#define BATCH 2
#define SEQ 2048
#define BH (BATCH * NUM_HEADS)   // 32

__device__ __forceinline__ unsigned short f2bf(float f) {
    unsigned u = __builtin_bit_cast(unsigned, f);
    unsigned r = u + 0x7FFFu + ((u >> 16) & 1u);   // round-to-nearest-even
    return (unsigned short)(r >> 16);
}

__device__ __forceinline__ v16bf cat8(v8bf lo, v8bf hi) {
    return __builtin_shufflevector(lo, hi, 0,1,2,3,4,5,6,7,8,9,10,11,12,13,14,15);
}

__device__ __forceinline__ v8f wmma_bf16(v16bf a, v16bf b, v8f c) {
    // 8 args: (neg_a, A, neg_b, B, c_mod, C, reuse_a, reuse_b)
    return __builtin_amdgcn_wmma_f32_16x16x32_bf16(false, a, false, b,
                                                   (short)0, c, false, false);
}

// Async copy: 16 bytes global -> LDS, tracked by ASYNCcnt (CDNA5 TDM-lite path).
// lds_off = byte address within the workgroup LDS allocation (generic-pointer
// low 32 bits, per ISA aperture rules).
__device__ __forceinline__ void async_copy_b128(unsigned lds_off, const void* gptr) {
    asm volatile("global_load_async_to_lds_b128 %0, %1, off"
                 :: "v"(lds_off), "v"(gptr) : "memory");
}
__device__ __forceinline__ void wait_asynccnt0() {
    asm volatile("s_wait_asynccnt 0x0" ::: "memory");
}
__device__ __forceinline__ unsigned lds_addr_of(const void* p) {
    return (unsigned)(unsigned long long)p;
}

// ---------------------------------------------------------------------------
// f32 -> bf16 cast
// ---------------------------------------------------------------------------
__global__ void cast_f32_to_bf16(const float* __restrict__ in,
                                 unsigned short* __restrict__ out, int n) {
    int i = blockIdx.x * blockDim.x + threadIdx.x;
    if (i < n) out[i] = f2bf(in[i]);
}

// ---------------------------------------------------------------------------
// WMMA GEMM:  C[m,n] = sum_k A[m,k] * W[n,k]   (i.e. X @ W.T)
// A: [M,K] bf16 row-major;  W: [N,K] bf16 row-major.
// Block 256 threads = 8 waves; tile 128x128x32; wave tile 64x32 (4x2 16x16).
// LDS double-buffered; tiles streamed in with GLOBAL_LOAD_ASYNC_TO_LDS_B128.
// mode 0: Cf[m*N+n] = acc + bias[n]           (f32)
// mode 1: Cb at [b,h,s,d] = ((b*H+h)*S+s)*64+d (bf16)   (Q, K)
// mode 2: Cb at [b,h,d,s] = ((b*H+h)*64+d)*S+s (bf16)   (V transposed)
// ---------------------------------------------------------------------------
__global__ __launch_bounds__(256) void gemm_xwT(
    const unsigned short* __restrict__ A,
    const unsigned short* __restrict__ W,
    float* __restrict__ Cf,
    unsigned short* __restrict__ Cb,
    const float* __restrict__ bias,
    int M, int N, int K, int mode)
{
    __shared__ __attribute__((aligned(32))) unsigned short As[2][128 * 32];
    __shared__ __attribute__((aligned(32))) unsigned short Bs[2][128 * 32];

    const int tid  = threadIdx.x;
    const int wave = tid >> 5;
    const int lane = tid & 31;
    const int lm   = lane & 15;
    const int lh   = lane >> 4;

    const int m0 = blockIdx.x * 128;
    const int n0 = blockIdx.y * 128;
    const int wm = (wave >> 2) * 64;   // wave M offset: 0 / 64
    const int wn = (wave & 3) * 32;    // wave N offset: 0..96

    v8f acc[4][2];
    #pragma unroll
    for (int i = 0; i < 4; ++i)
        #pragma unroll
        for (int j = 0; j < 2; ++j)
            acc[i][j] = (v8f){0.f,0.f,0.f,0.f,0.f,0.f,0.f,0.f};

    const int srow = tid >> 1;          // 0..127
    const int sseg = (tid & 1) * 16;    // 0 / 16 halves

    // issue the 4 async b128 copies staging one (A,B) K-tile into buffer `buf`
    auto issue_tile = [&](int buf, int k0) {
        const unsigned short* ga = A + (size_t)(m0 + srow) * K + k0 + sseg;
        unsigned la = lds_addr_of(&As[buf][srow * 32 + sseg]);
        async_copy_b128(la,      ga);
        async_copy_b128(la + 16, ga + 8);
        const unsigned short* gb = W + (size_t)(n0 + srow) * K + k0 + sseg;
        unsigned lb = lds_addr_of(&Bs[buf][srow * 32 + sseg]);
        async_copy_b128(lb,      gb);
        async_copy_b128(lb + 16, gb + 8);
    };

    issue_tile(0, 0);
    int buf = 0;

    for (int k0 = 0; k0 < K; k0 += 32) {
        wait_asynccnt0();       // own async copies landed in LDS
        __syncthreads();        // all waves' copies landed; prev buffer free
        if (k0 + 32 < K)
            issue_tile(buf ^ 1, k0 + 32);   // stream next tile while computing

        // ---- A fragments (lane=M, half-wave h holds K 16h..16h+15) ----
        const unsigned short* as = As[buf];
        const unsigned short* bs = Bs[buf];
        v16bf afr[4];
        #pragma unroll
        for (int i = 0; i < 4; ++i) {
            const int m = wm + i * 16 + lm;
            v8bf lo = *(const v8bf*)(as + m * 32 + lh * 16);       // K 16h..16h+7
            v8bf hi = *(const v8bf*)(as + m * 32 + lh * 16 + 8);   // K 16h+8..16h+15
            afr[i] = cat8(lo, hi);
        }
        // ---- B fragments (lane=N, half-wave selects K 0-15 / 16-31) ----
        #pragma unroll
        for (int j = 0; j < 2; ++j) {
            const int n = wn + j * 16 + lm;
            v16bf bfr = *(const v16bf*)(bs + n * 32 + lh * 16);
            #pragma unroll
            for (int i = 0; i < 4; ++i)
                acc[i][j] = wmma_bf16(afr[i], bfr, acc[i][j]);
        }
        buf ^= 1;
    }

    // ---- epilogue ----
    #pragma unroll
    for (int i = 0; i < 4; ++i) {
        #pragma unroll
        for (int j = 0; j < 2; ++j) {
            const int gn = n0 + wn + j * 16 + lm;
            #pragma unroll
            for (int v = 0; v < 8; ++v) {
                const int gm = m0 + wm + i * 16 + v + 8 * lh;
                const float val = acc[i][j][v];
                if (mode == 0) {
                    Cf[(size_t)gm * N + gn] = val + bias[gn];
                } else {
                    const int b = gm >> 11, s = gm & (SEQ - 1);     // SEQ = 2048
                    const int h = gn >> 6,  d = gn & (D_K - 1);
                    size_t idx;
                    if (mode == 1)
                        idx = ((size_t)(b * NUM_HEADS + h) * SEQ + s) * D_K + d;
                    else
                        idx = ((size_t)(b * NUM_HEADS + h) * D_K + d) * SEQ + s;
                    Cb[idx] = f2bf(val);
                }
            }
        }
    }
}

// ---------------------------------------------------------------------------
// Flash-attention (causal), one wave per 16-query tile.
// Qb, Kb: [BH, S, 64] bf16 ; Vt: [BH, 64, S] bf16 (transposed)
// Ob: [B, S, 1024] bf16  (attn output regrouped for the output projection)
// ---------------------------------------------------------------------------
__global__ __launch_bounds__(256) void attn_flash(
    const unsigned short* __restrict__ Qb,
    const unsigned short* __restrict__ Kb,
    const unsigned short* __restrict__ Vt,
    unsigned short* __restrict__ Ob)
{
    __shared__ __attribute__((aligned(32))) unsigned short Ps[8][16][32]; // per-wave P

    const int wave = threadIdx.x >> 5;
    const int lane = threadIdx.x & 31;
    const int lm   = lane & 15;
    const int lh   = lane >> 4;

    const int t  = blockIdx.x * 8 + wave;      // 0 .. 4095 query tiles
    const int bh = t >> 7;                     // / (SEQ/16 = 128)
    const int qb = (t & 127) << 4;

    const unsigned short* Q  = Qb + (size_t)bh * SEQ * D_K;
    const unsigned short* Kp = Kb + (size_t)bh * SEQ * D_K;
    const unsigned short* Vp = Vt + (size_t)bh * D_K * SEQ;

    // Q A-fragments over dk = 64 (two K=32 chunks)
    v16bf qf[2];
    #pragma unroll
    for (int kc = 0; kc < 2; ++kc) {
        const unsigned short* qr = Q + (size_t)(qb + lm) * D_K + kc * 32 + lh * 16;
        qf[kc] = cat8(*(const v8bf*)qr, *(const v8bf*)(qr + 8));
    }

    v8f acc[4];
    #pragma unroll
    for (int j = 0; j < 4; ++j)
        acc[j] = (v8f){0.f,0.f,0.f,0.f,0.f,0.f,0.f,0.f};
    float ms[8], ls[8];
    #pragma unroll
    for (int v = 0; v < 8; ++v) { ms[v] = -3.0e38f; ls[v] = 0.f; }

    const float scale = 0.125f;                // 1/sqrt(64)

    for (int kb = 0; kb < qb + 16; kb += 32) {
        // ---- scores: 16x32 tile, two 16x16 WMMA pairs over dk ----
        v8f sc[2];
        #pragma unroll
        for (int nt = 0; nt < 2; ++nt) {
            const unsigned short* kr =
                Kp + (size_t)(kb + nt * 16 + lm) * D_K + lh * 16;
            v16bf kf0 = *(const v16bf*)kr;          // dk 0..31 group
            v16bf kf1 = *(const v16bf*)(kr + 32);   // dk 32..63 group
            v8f z = (v8f){0.f,0.f,0.f,0.f,0.f,0.f,0.f,0.f};
            z = wmma_bf16(qf[0], kf0, z);
            z = wmma_bf16(qf[1], kf1, z);
            sc[nt] = z;
        }

        // ---- online softmax per row (8 rows per half-wave) ----
        #pragma unroll
        for (int v = 0; v < 8; ++v) {
            const int qi = qb + v + 8 * lh;
            float s0 = sc[0][v] * scale;
            float s1 = sc[1][v] * scale;
            if (kb + lm      > qi) s0 = -1.0e30f;   // causal mask
            if (kb + 16 + lm > qi) s1 = -1.0e30f;
            float mx = fmaxf(s0, s1);
            #pragma unroll
            for (int d = 1; d < 16; d <<= 1)
                mx = fmaxf(mx, __shfl_xor(mx, d, 32));
            const float mnew  = fmaxf(ms[v], mx);
            const float alpha = __expf(ms[v] - mnew);
            const float p0 = __expf(s0 - mnew);
            const float p1 = __expf(s1 - mnew);
            float rs = p0 + p1;
            #pragma unroll
            for (int d = 1; d < 16; d <<= 1)
                rs += __shfl_xor(rs, d, 32);
            ls[v] = ls[v] * alpha + rs;
            ms[v] = mnew;
            #pragma unroll
            for (int j = 0; j < 4; ++j)
                acc[j][v] *= alpha;
            const int m = v + 8 * lh;
            Ps[wave][m][lm]      = f2bf(p0);        // C-layout -> LDS row-major
            Ps[wave][m][16 + lm] = f2bf(p1);
        }
        asm volatile("s_wait_dscnt 0" ::: "memory");

        // ---- P as A-fragment (16x32, K = 32 keys) ----
        v16bf pf = cat8(*(const v8bf*)(&Ps[wave][lm][lh * 16]),
                        *(const v8bf*)(&Ps[wave][lm][lh * 16 + 8]));

        // ---- O += P @ V : 4 N-tiles of 16 over dk ----
        #pragma unroll
        for (int j = 0; j < 4; ++j) {
            const unsigned short* vr =
                Vp + (size_t)(j * 16 + lm) * SEQ + kb + lh * 16;
            v16bf vf = *(const v16bf*)vr;
            acc[j] = wmma_bf16(pf, vf, acc[j]);
        }
    }

    // ---- finalize: O /= l, write bf16 to [B, S, D_MODEL] ----
    const int b = bh >> 4, h = bh & (NUM_HEADS - 1);
    #pragma unroll
    for (int v = 0; v < 8; ++v) {
        const float inv = 1.0f / ls[v];
        const int s = qb + v + 8 * lh;
        const size_t base = ((size_t)b * SEQ + s) * D_MODEL + h * D_K;
        #pragma unroll
        for (int j = 0; j < 4; ++j)
            Ob[base + j * 16 + lm] = f2bf(acc[j][v] * inv);
    }
}

// ---------------------------------------------------------------------------
// Launch
// ---------------------------------------------------------------------------
extern "C" void kernel_launch(void* const* d_in, const int* in_sizes, int n_in,
                              void* d_out, int out_size, void* d_ws, size_t ws_size,
                              hipStream_t stream) {
    (void)in_sizes; (void)n_in; (void)out_size; (void)ws_size;

    const float* x   = (const float*)d_in[0];
    // d_in[1] = mask (bool) : causal mask computed analytically, unused
    const float* W_q = (const float*)d_in[2];
    const float* W_k = (const float*)d_in[3];
    const float* W_v = (const float*)d_in[4];
    const float* W_o = (const float*)d_in[5];
    const float* b_o = (const float*)d_in[6];
    float* out = (float*)d_out;

    char* ws = (char*)d_ws;
    const size_t MB = 1024 * 1024;
    unsigned short* Xb   = (unsigned short*)(ws +  0 * MB);  // [4096,1024]  8MB
    unsigned short* Wqb  = (unsigned short*)(ws +  8 * MB);  // [1024,1024]  2MB
    unsigned short* Wkb  = (unsigned short*)(ws + 10 * MB);
    unsigned short* Wvb  = (unsigned short*)(ws + 12 * MB);
    unsigned short* Wob  = (unsigned short*)(ws + 14 * MB);
    unsigned short* Qbuf = (unsigned short*)(ws + 16 * MB);  // [32,2048,64] 8MB
    unsigned short* Kbuf = (unsigned short*)(ws + 24 * MB);
    unsigned short* Vtbf = (unsigned short*)(ws + 32 * MB);  // [32,64,2048] 8MB
    unsigned short* Attn = (unsigned short*)(ws + 40 * MB);  // [4096,1024]  8MB

    const int M = BATCH * SEQ;   // 4096
    const int N = D_MODEL;       // 1024
    const int K = D_MODEL;       // 1024

    // casts
    {
        const int nx = M * K;
        cast_f32_to_bf16<<<nx / 256, 256, 0, stream>>>(x, Xb, nx);
        const int nw = N * K;
        cast_f32_to_bf16<<<nw / 256, 256, 0, stream>>>(W_q, Wqb, nw);
        cast_f32_to_bf16<<<nw / 256, 256, 0, stream>>>(W_k, Wkb, nw);
        cast_f32_to_bf16<<<nw / 256, 256, 0, stream>>>(W_v, Wvb, nw);
        cast_f32_to_bf16<<<nw / 256, 256, 0, stream>>>(W_o, Wob, nw);
    }

    dim3 ggrid(M / 128, N / 128);   // (32, 8)
    // Q, K projections -> [B,H,S,dk]
    gemm_xwT<<<ggrid, 256, 0, stream>>>(Xb, Wqb, nullptr, Qbuf, nullptr, M, N, K, 1);
    gemm_xwT<<<ggrid, 256, 0, stream>>>(Xb, Wkb, nullptr, Kbuf, nullptr, M, N, K, 1);
    // V projection -> transposed [B,H,dk,S]
    gemm_xwT<<<ggrid, 256, 0, stream>>>(Xb, Wvb, nullptr, Vtbf, nullptr, M, N, K, 2);

    // flash attention: 4096 query tiles, 8 waves/block
    attn_flash<<<512, 256, 0, stream>>>(Qbuf, Kbuf, Vtbf, Attn);

    // output projection + bias -> f32
    gemm_xwT<<<ggrid, 256, 0, stream>>>(Attn, Wob, out, nullptr, b_o, M, N, K, 0);
}
